// CrossWinAttention_44006234915574
// MI455X (gfx1250) — matrix operand order repaired and checked by reading, not compiled, not run on hardware
//
#include <hip/hip_runtime.h>
#include <hip/hip_bf16.h>
#include <cstddef>

typedef __attribute__((ext_vector_type(16))) _Float16 v16h;
typedef __attribute__((ext_vector_type(8)))  float    v8f;

#define DIMC 128
#define TWIN 384
#define DHEADC 32

__device__ __forceinline__ v8f wmma_f16(v16h a, v16h b, v8f c) {
    return __builtin_amdgcn_wmma_f32_16x16x32_f16(false, a, false, b, (short)0, c, false, false);
}

// A-fragment 16x32 f16 from row-major LDS matrix [rows][ld] at (row0, k0).
// Contiguous k order used here: lanes 0-15: M=lane, K = 0..15; lanes 16-31: M=lane-16, K = 16..31.
__device__ __forceinline__ v16h load_a_frag(const _Float16* base, int ld, int row0, int k0) {
    const int lid = threadIdx.x & 31;
    const int m   = lid & 15;
    const int kb  = (lid >> 4) << 4;
    const _Float16* p = base + (row0 + m) * ld + k0 + kb;
    union { v16h v; float4 f[2]; } u;
    u.f[0] = *(const float4*)(p);        // K = kb .. kb+7
    u.f[1] = *(const float4*)(p + 8);    // K = kb+8 .. kb+15
    return u.v;
}

// B-fragment 32x16 f16 from a TRANSPOSED row-major LDS matrix Bt[n][k] with leading dim ld.
// Contiguous k order used here: lanes 0-15: N=lane, K=0..15; lanes 16-31: N=lane-16, K=16..31.
__device__ __forceinline__ v16h load_bt_frag(const _Float16* bt, int ld, int n0, int k0) {
    const int lid = threadIdx.x & 31;
    const int n   = lid & 15;
    const int ks  = k0 + ((lid >> 4) << 4);
    const _Float16* p = bt + (n0 + n) * ld + ks;
    union { v16h v; float4 f[2]; } u;
    u.f[0] = *(const float4*)(p);        // K = ks .. ks+7
    u.f[1] = *(const float4*)(p + 8);    // K = ks+8 .. ks+15
    return u.v;
}

// ---------------------------------------------------------------------------
// Kernel 1 helper: LN + projection for ONE input stream (direct pointers so
// address-space inference yields global_* (not flat_*) memory ops).
// ---------------------------------------------------------------------------
__device__ __forceinline__ void ln_proj_one(
    const float* __restrict__ inp, const float* __restrict__ W,
    const float* __restrict__ bias, const float* __restrict__ lnw,
    const float* __restrict__ lnb, const float* __restrict__ head_gate,
    bool isQ, _Float16* __restrict__ outp,
    _Float16* sW, _Float16* sA,
    float* sBias, float* sGate, float* sLnW, float* sLnB,
    int b, int l, int x, int y, int tid, int wave, int lid)
{
    __syncthreads();   // previous use of sW/sA complete
    for (int i = tid; i < DIMC * DIMC; i += 256) {
        int kk = i >> 7, nn = i & 127;              // w[k][n] row-major (in,out)
        sW[nn * DIMC + kk] = (_Float16)W[i];
    }
    if (tid < DIMC) {
        const float scale = 0.1767766952966369f;    // 32^-0.5
        sBias[tid] = bias[tid];
        sGate[tid] = isQ ? scale * head_gate[tid >> 5] : 1.0f;
        sLnW[tid]  = lnw[tid];
        sLnB[tid]  = lnb[tid];
    }

    for (int tile = 0; tile < 3; ++tile) {
        __syncthreads();   // prior GEMM/staging done before overwriting sA
        // --- LayerNorm 128 tokens: one wave per token, 4 dims/lane ---
        for (int i = 0; i < 16; ++i) {
            int tl = wave * 16 + i;
            int t  = tile * 128 + tl;
            int n  = t >> 6, s = t & 63;
            const float* src = inp +
                ((size_t)(((b * 6 + n) * 8 + x) * 8 + y) * 64 + s) * DIMC;
            float4 xv = *(const float4*)(src + lid * 4);
            float sum = xv.x + xv.y + xv.z + xv.w;
            float sq  = xv.x * xv.x + xv.y * xv.y + xv.z * xv.z + xv.w * xv.w;
            #pragma unroll
            for (int msk = 16; msk >= 1; msk >>= 1) {
                sum += __shfl_xor(sum, msk, 32);
                sq  += __shfl_xor(sq,  msk, 32);
            }
            float mu   = sum * (1.0f / 128.0f);
            float var  = sq * (1.0f / 128.0f) - mu * mu;
            float rstd = rsqrtf(var + 1e-5f);
            int d0 = lid * 4;
            sA[tl * DIMC + d0 + 0] = (_Float16)((xv.x - mu) * rstd * sLnW[d0 + 0] + sLnB[d0 + 0]);
            sA[tl * DIMC + d0 + 1] = (_Float16)((xv.y - mu) * rstd * sLnW[d0 + 1] + sLnB[d0 + 1]);
            sA[tl * DIMC + d0 + 2] = (_Float16)((xv.z - mu) * rstd * sLnW[d0 + 2] + sLnB[d0 + 2]);
            sA[tl * DIMC + d0 + 3] = (_Float16)((xv.w - mu) * rstd * sLnW[d0 + 3] + sLnB[d0 + 3]);
        }
        __syncthreads();
        // --- GEMM 128x128x128: wave owns m-tile = wave (16 rows) ---
        v16h a0 = load_a_frag(sA, DIMC, wave * 16, 0);
        v16h a1 = load_a_frag(sA, DIMC, wave * 16, 32);
        v16h a2 = load_a_frag(sA, DIMC, wave * 16, 64);
        v16h a3 = load_a_frag(sA, DIMC, wave * 16, 96);
        const int mbase = (lid < 16) ? 0 : 8;
        // After the a-frag ds_loads complete (consumed by WMMA), this wave's
        // own 16 rows of sA are dead -> reuse as coalescing staging buffer.
        _Float16* stage = sA + (size_t)wave * 16 * DIMC;
        for (int nt = 0; nt < 8; ++nt) {
            v8f c = {};
            c = wmma_f16(a0, load_bt_frag(sW, DIMC, nt * 16, 0),  c);
            c = wmma_f16(a1, load_bt_frag(sW, DIMC, nt * 16, 32), c);
            c = wmma_f16(a2, load_bt_frag(sW, DIMC, nt * 16, 64), c);
            c = wmma_f16(a3, load_bt_frag(sW, DIMC, nt * 16, 96), c);
            int nn = nt * 16 + (lid & 15);
            float bb = sBias[nn], gg = sGate[nn];
            #pragma unroll
            for (int r = 0; r < 8; ++r)
                stage[(r + mbase) * DIMC + nn] = (_Float16)((c[r] + bb) * gg);
        }
        // same-wave LDS ops are in-order: staging writes visible to reads below
        #pragma unroll
        for (int it = 0; it < 8; ++it) {
            int i  = it * 32 + lid;          // float4 index within 16x128 tile
            int hw = i * 8;                  // half index
            int rr = hw >> 7;                // local row 0..15
            int col = hw & 127;
            int h = col >> 5, dh0 = col & 31;
            int t = tile * 128 + wave * 16 + rr;
            _Float16* dst = outp +
                ((size_t)((b * 4 + h) * 64 + l) * TWIN + t) * DHEADC + dh0;
            *(float4*)dst = *(const float4*)(stage + hw);
        }
    }
}

// ---------------------------------------------------------------------------
// Kernel 1: fused LayerNorm + {Q,K,V} projection, fold (scale*head_gate) into Q.
// Grid: B*64 blocks (one per (b, window)); block = 256 threads (8 waves).
// Outputs f16 in layout ws[((b*4+h)*64 + l)*384 + t][32].
// ---------------------------------------------------------------------------
__global__ __launch_bounds__(256) void k_ln_qkv(
    const float* __restrict__ q, const float* __restrict__ k, const float* __restrict__ v,
    const float* __restrict__ head_gate,
    const float* __restrict__ lnqw, const float* __restrict__ lnqb,
    const float* __restrict__ lnkw, const float* __restrict__ lnkb,
    const float* __restrict__ lnvw, const float* __restrict__ lnvb,
    const float* __restrict__ wq, const float* __restrict__ bq,
    const float* __restrict__ wk, const float* __restrict__ bk,
    const float* __restrict__ wv, const float* __restrict__ bv,
    _Float16* __restrict__ qws, _Float16* __restrict__ kws, _Float16* __restrict__ vws)
{
    __shared__ __align__(16) _Float16 sW[DIMC * DIMC];   // weight, transposed Wt[n][k]
    __shared__ __align__(16) _Float16 sA[128 * DIMC];    // LN'd token tile / staging
    __shared__ float sBias[DIMC], sGate[DIMC], sLnW[DIMC], sLnB[DIMC];

    const int blk = blockIdx.x;          // b*64 + l
    const int b = blk >> 6;
    const int l = blk & 63;
    const int x = l >> 3, y = l & 7;
    const int tid = threadIdx.x;
    const int wave = tid >> 5, lid = tid & 31;

    ln_proj_one(q, wq, bq, lnqw, lnqb, head_gate, true,  qws,
                sW, sA, sBias, sGate, sLnW, sLnB, b, l, x, y, tid, wave, lid);
    ln_proj_one(k, wk, bk, lnkw, lnkb, head_gate, false, kws,
                sW, sA, sBias, sGate, sLnW, sLnB, b, l, x, y, tid, wave, lid);
    ln_proj_one(v, wv, bv, lnvw, lnvb, head_gate, false, vws,
                sW, sA, sBias, sGate, sLnW, sLnB, b, l, x, y, tid, wave, lid);
}

// ---------------------------------------------------------------------------
// Kernel 2: attention per (b, head, window). softmax(Q K^T) V, online softmax.
// Grid: 512 blocks; block 256 (8 waves). Each wave handles 3 tiles of 16 query rows.
// ---------------------------------------------------------------------------
__global__ __launch_bounds__(256) void k_attn(
    const _Float16* __restrict__ qws, const _Float16* __restrict__ kws,
    const _Float16* __restrict__ vws, _Float16* __restrict__ aws)
{
    __shared__ __align__(16) _Float16 sQ[TWIN * DHEADC];     // 24 KB
    __shared__ __align__(16) _Float16 sK[TWIN * DHEADC];     // 24 KB
    __shared__ __align__(16) _Float16 sVt[DHEADC * TWIN];    // 24 KB (transposed)
    __shared__ __align__(16) _Float16 sP[8][16 * 32];        // per-wave P / O staging

    const int blk = blockIdx.x;               // (b*4+h)*64 + l
    const int b = blk >> 8;
    const int h = (blk >> 6) & 3;
    const int l = blk & 63;
    const size_t base = (size_t)blk * TWIN * DHEADC;
    const int tid = threadIdx.x, wave = tid >> 5, lid = tid & 31;

    for (int i = tid; i < TWIN * DHEADC / 8; i += 256) {
        ((float4*)sQ)[i] = ((const float4*)(qws + base))[i];
        ((float4*)sK)[i] = ((const float4*)(kws + base))[i];
    }
    for (int i = tid; i < TWIN * DHEADC; i += 256) {
        int t = i >> 5, dh = i & 31;
        sVt[dh * TWIN + t] = vws[base + i];
    }
    __syncthreads();

    const int mbase = (lid < 16) ? 0 : 8;
    for (int qt = wave; qt < TWIN / 16; qt += 8) {
        v16h aQ = load_a_frag(sQ, DHEADC, qt * 16, 0);
        v8f o0 = {}, o1 = {};
        float mrow[8], lrow[8];
        #pragma unroll
        for (int r = 0; r < 8; ++r) { mrow[r] = -1e30f; lrow[r] = 0.0f; }
        _Float16* sPw = &sP[wave][0];

        for (int c = 0; c < TWIN / 32; ++c) {
            // scores: 16 q-rows x 32 k-tokens (sK is row-major [t][dh] == Bt[n][k])
            v8f s0 = {}, s1 = {};
            s0 = wmma_f16(aQ, load_bt_frag(sK, DHEADC, c * 32,      0), s0);
            s1 = wmma_f16(aQ, load_bt_frag(sK, DHEADC, c * 32 + 16, 0), s1);
            #pragma unroll
            for (int r = 0; r < 8; ++r) {
                float v0 = s0[r], v1 = s1[r];
                float mx = fmaxf(v0, v1);
                #pragma unroll
                for (int msk = 8; msk >= 1; msk >>= 1)
                    mx = fmaxf(mx, __shfl_xor(mx, msk, 32));
                float mnew  = fmaxf(mrow[r], mx);
                float alpha = __expf(mrow[r] - mnew);
                float p0 = __expf(v0 - mnew);
                float p1 = __expf(v1 - mnew);
                float ps = p0 + p1;
                #pragma unroll
                for (int msk = 8; msk >= 1; msk >>= 1)
                    ps += __shfl_xor(ps, msk, 32);
                lrow[r] = lrow[r] * alpha + ps;
                mrow[r] = mnew;
                o0[r] *= alpha;
                o1[r] *= alpha;
                int m  = r + mbase;
                int kc = lid & 15;
                sPw[m * 32 + kc]      = (_Float16)p0;
                sPw[m * 32 + kc + 16] = (_Float16)p1;
            }
            // P (16x32) x V-chunk (32x32): sVt[n][k] is the transposed B
            v16h aP = load_a_frag(sPw, 32, 0, 0);
            o0 = wmma_f16(aP, load_bt_frag(sVt, TWIN, 0,  c * 32), o0);
            o1 = wmma_f16(aP, load_bt_frag(sVt, TWIN, 16, c * 32), o1);
        }
        // normalize, stage 16x32 O tile in per-wave LDS, then coalesced stores
        int nn = lid & 15;
        #pragma unroll
        for (int r = 0; r < 8; ++r) {
            float inv = 1.0f / lrow[r];
            sPw[(r + mbase) * 32 + nn]      = (_Float16)(o0[r] * inv);
            sPw[(r + mbase) * 32 + nn + 16] = (_Float16)(o1[r] * inv);
        }
        #pragma unroll
        for (int it = 0; it < 2; ++it) {
            int i  = it * 32 + lid;          // float4 index within 16x32 tile
            int hw = i * 8;                  // half index
            int rr = hw >> 5;                // local row 0..15
            int dh0 = hw & 31;
            int t = qt * 16 + rr;
            _Float16* dst = aws +
                ((size_t)((b * 64 + l) * TWIN + t)) * 128 + h * 32 + dh0;
            *(float4*)dst = *(const float4*)(sPw + hw);
        }
    }
}

// ---------------------------------------------------------------------------
// Kernel 3: output projection a @ wp + bp -> z (f32), windows layout.
// ---------------------------------------------------------------------------
__global__ __launch_bounds__(256) void k_proj_out(
    const _Float16* __restrict__ aws, const float* __restrict__ wp,
    const float* __restrict__ bp, float* __restrict__ zws)
{
    __shared__ __align__(16) _Float16 sW[DIMC * DIMC];
    __shared__ __align__(16) _Float16 sA[128 * DIMC];
    __shared__ float sBias[DIMC];

    const int blk = blockIdx.x;     // b*64 + l
    const int tid = threadIdx.x, wave = tid >> 5, lid = tid & 31;

    for (int i = tid; i < DIMC * DIMC; i += 256) {
        int kk = i >> 7, nn = i & 127;
        sW[nn * DIMC + kk] = (_Float16)wp[i];
    }
    if (tid < DIMC) sBias[tid] = bp[tid];

    const size_t wbase = (size_t)blk * TWIN * DIMC;
    const int mbase = (lid < 16) ? 0 : 8;
    for (int tile = 0; tile < 3; ++tile) {
        __syncthreads();
        for (int i = tid; i < 128 * DIMC / 8; i += 256)
            ((float4*)sA)[i] = ((const float4*)(aws + wbase + (size_t)tile * 128 * DIMC))[i];
        __syncthreads();
        v16h a0 = load_a_frag(sA, DIMC, wave * 16, 0);
        v16h a1 = load_a_frag(sA, DIMC, wave * 16, 32);
        v16h a2 = load_a_frag(sA, DIMC, wave * 16, 64);
        v16h a3 = load_a_frag(sA, DIMC, wave * 16, 96);
        for (int nt = 0; nt < 8; ++nt) {
            v8f c = {};
            c = wmma_f16(a0, load_bt_frag(sW, DIMC, nt * 16, 0),  c);
            c = wmma_f16(a1, load_bt_frag(sW, DIMC, nt * 16, 32), c);
            c = wmma_f16(a2, load_bt_frag(sW, DIMC, nt * 16, 64), c);
            c = wmma_f16(a3, load_bt_frag(sW, DIMC, nt * 16, 96), c);
            int nn = nt * 16 + (lid & 15);
            float bias = sBias[nn];
            float* dst = zws + wbase + (size_t)(tile * 128 + wave * 16 + mbase) * DIMC + nn;
            #pragma unroll
            for (int r = 0; r < 8; ++r)
                dst[(size_t)r * DIMC] = c[r] + bias;
        }
    }
}

// ---------------------------------------------------------------------------
// Kernel 4: mean over agents n (6) + skip connection.
// ---------------------------------------------------------------------------
__global__ void k_mean_skip(const float* __restrict__ zws,
                            const float* __restrict__ skip,
                            float* __restrict__ out)
{
    int idx = blockIdx.x * blockDim.x + threadIdx.x;
    if (idx >= 2 * 8 * 8 * 8 * 8 * 128) return;
    int d = idx & 127;
    int s = (idx >> 7) & 63;
    int y = (idx >> 13) & 7;
    int x = (idx >> 16) & 7;
    int b = idx >> 19;
    int l = x * 8 + y;
    size_t base = (size_t)(b * 64 + l) * TWIN * 128;
    float acc = 0.0f;
    #pragma unroll
    for (int n = 0; n < 6; ++n)
        acc += zws[base + (size_t)(n * 64 + s) * 128 + d];
    out[idx] = acc * (1.0f / 6.0f) + skip[idx];
}

// ---------------------------------------------------------------------------
extern "C" void kernel_launch(void* const* d_in, const int* in_sizes, int n_in,
                              void* d_out, int out_size, void* d_ws, size_t ws_size,
                              hipStream_t stream)
{
    const float* q        = (const float*)d_in[0];
    const float* k        = (const float*)d_in[1];
    const float* v        = (const float*)d_in[2];
    const float* skip     = (const float*)d_in[3];
    const float* head_g   = (const float*)d_in[4];
    const float* ln_q_w   = (const float*)d_in[5];
    const float* ln_q_b   = (const float*)d_in[6];
    const float* ln_k_w   = (const float*)d_in[7];
    const float* ln_k_b   = (const float*)d_in[8];
    const float* ln_v_w   = (const float*)d_in[9];
    const float* ln_v_b   = (const float*)d_in[10];
    const float* wq       = (const float*)d_in[11];
    const float* bq       = (const float*)d_in[12];
    const float* wk       = (const float*)d_in[13];
    const float* bk       = (const float*)d_in[14];
    const float* wv       = (const float*)d_in[15];
    const float* bv       = (const float*)d_in[16];
    const float* wp       = (const float*)d_in[17];
    const float* bp       = (const float*)d_in[18];
    float* out = (float*)d_out;

    char* ws = (char*)d_ws;
    const size_t HALF_QKV = (size_t)2 * 4 * 64 * TWIN * DHEADC * sizeof(_Float16); // 12,582,912
    _Float16* qws = (_Float16*)(ws + 0);
    _Float16* kws = (_Float16*)(ws + HALF_QKV);
    _Float16* vws = (_Float16*)(ws + 2 * HALF_QKV);
    _Float16* aws = (_Float16*)(ws + 3 * HALF_QKV);
    float*    zws = (float*)(ws + 0);   // overlays q/k ws (dead by kernel 3)

    k_ln_qkv<<<128, 256, 0, stream>>>(q, k, v, head_g,
                                      ln_q_w, ln_q_b, ln_k_w, ln_k_b, ln_v_w, ln_v_b,
                                      wq, bq, wk, bk, wv, bv, qws, kws, vws);
    k_attn<<<512, 256, 0, stream>>>(qws, kws, vws, aws);
    k_proj_out<<<128, 256, 0, stream>>>(aws, wp, bp, zws);
    k_mean_skip<<<(2 * 8 * 8 * 8 * 8 * 128 + 255) / 256, 256, 0, stream>>>(zws, skip, out);
}
